// AbsHMM_30459908063825
// MI455X (gfx1250) — matrix operand order repaired and checked by reading, not compiled, not run on hardware
//
#include <hip/hip_runtime.h>
#include <hip/hip_bf16.h>

// HMM forward (B=32, T=4096, C=64) restructured as chunked products of
// per-step transition matrices M_t = A * diag(exp(x_t)) in linear space with
// exact power-of-2 rescaling.  Phase 1: 32 batches x 32 chunks of 128 factors,
// each chunk composed by a 4-wave workgroup doing 64x64x64 f16 WMMA matmuls
// in LDS.  Phase 2: per-batch sequential combine of 32 chunk matrices.
//
// CDNA5 paths: v_wmma_f32_16x16x32_f16 core; async global<->LDS staging via
// GLOBAL_LOAD_ASYNC_TO_LDS_B128 / GLOBAL_STORE_ASYNC_FROM_LDS_B128 with
// s_wait_asynccnt (guarded; falls back to VGPR copies if undeclared).

typedef _Float16 v16h __attribute__((ext_vector_type(16)));
typedef float    v8f  __attribute__((ext_vector_type(8)));
typedef int      v4i  __attribute__((vector_size(16)));   // builtin param type

union Frag16 { v16h h; uint4 q[2]; };

#define HMM_B   32
#define HMM_T   4096
#define HMM_C   64
#define LCH     128          // chunk length (T = NCH * LCH)
#define NCH     32           // chunks per batch
#define PSTR    72           // f16 elements per padded LDS row (36 dwords)

#define LN2F    0.6931471805599453f

#define AS_GLOBAL __attribute__((address_space(1)))
#define AS_LDS    __attribute__((address_space(3)))

#if defined(__has_builtin)
#if __has_builtin(__builtin_amdgcn_global_load_async_to_lds_b128) && \
    __has_builtin(__builtin_amdgcn_s_wait_asynccnt)
#define HMM_ASYNC_LD 1
#endif
#if __has_builtin(__builtin_amdgcn_global_store_async_from_lds_b128) && \
    __has_builtin(__builtin_amdgcn_s_wait_asynccnt)
#define HMM_ASYNC_ST 1
#endif
#endif
#ifndef HMM_ASYNC_LD
#define HMM_ASYNC_LD 0
#endif
#ifndef HMM_ASYNC_ST
#define HMM_ASYNC_ST 0
#endif

// ---------------------------------------------------------------------------
// Kernel 0: A = softmax(log_A, axis=1) as f16;  pi = softmax(log_pi) as f32.
// ---------------------------------------------------------------------------
__global__ __launch_bounds__(64) void hmm_prep(const float* __restrict__ logA,
                                               const float* __restrict__ logpi,
                                               _Float16* __restrict__ A16,
                                               float* __restrict__ pi) {
    const int i = threadIdx.x;   // 0..63, one row per thread
    float m = -INFINITY;
    for (int j = 0; j < HMM_C; ++j) m = fmaxf(m, logA[i * HMM_C + j]);
    float s = 0.f;
    for (int j = 0; j < HMM_C; ++j) s += __expf(logA[i * HMM_C + j] - m);
    const float inv = 1.0f / s;
    for (int j = 0; j < HMM_C; ++j)
        A16[i * HMM_C + j] = (_Float16)(__expf(logA[i * HMM_C + j] - m) * inv);

    // pi (redundantly per thread; trivial)
    float mp = -INFINITY;
    for (int j = 0; j < HMM_C; ++j) mp = fmaxf(mp, logpi[j]);
    float sp = 0.f;
    for (int j = 0; j < HMM_C; ++j) sp += __expf(logpi[j] - mp);
    pi[i] = __expf(logpi[i] - mp) / sp;
}

// ---------------------------------------------------------------------------
// Kernel 1: compose one chunk of LCH transition factors per (chunk, batch).
//   P <- P @ (A * diag(exp(x_t))) , 127 times, f16 operands / f32 accum WMMA.
//   grid = (NCH, B), block = 128 threads (4 waves, wave w owns M rows 16w..).
// ---------------------------------------------------------------------------
__global__ __launch_bounds__(128) void hmm_chunks(const float* __restrict__ x,
                                                  const _Float16* __restrict__ A16,
                                                  const float* __restrict__ pi,
                                                  _Float16* __restrict__ Pout,
                                                  float* __restrict__ logSout) {
    __shared__ __align__(16) _Float16 As[HMM_C * PSTR];   // padded A, row-major
    __shared__ __align__(16) _Float16 Ps[HMM_C * PSTR];   // padded P, row-major
    __shared__ __align__(16) float    xs[LCH * HMM_C];    // chunk emissions
    __shared__ float wmax[4];

    const int tid  = threadIdx.x;
    const int c    = blockIdx.x;
    const int b    = blockIdx.y;
    const int lane = tid & 31;
    const int w    = tid >> 5;
    const int l15  = lane & 15;
    const int hi   = lane >> 4;

    // ---- stage A (padded) and the chunk's emissions into LDS ----
    const float* gxp = x + ((size_t)b * HMM_T + (size_t)c * LCH) * HMM_C;
#if HMM_ASYNC_LD
    {
        // A: 4096 f16 = 512 x b128, scattered into padded 144B rows.
        for (int idx = tid; idx < 512; idx += 128) {
            const int i = idx >> 3, q = idx & 7;
            __builtin_amdgcn_global_load_async_to_lds_b128(
                (AS_GLOBAL v4i*)((AS_GLOBAL char*)A16 + (size_t)idx * 16),
                (AS_LDS v4i*)((AS_LDS char*)As + (size_t)i * (PSTR * 2) + (size_t)q * 16),
                0, 0);
        }
        // emissions: LCH*64 f32 = 2048 x b128, contiguous.
        for (int idx = tid; idx < LCH * 16; idx += 128) {
            __builtin_amdgcn_global_load_async_to_lds_b128(
                (AS_GLOBAL v4i*)((AS_GLOBAL char*)gxp + (size_t)idx * 16),
                (AS_LDS v4i*)((AS_LDS char*)xs + (size_t)idx * 16),
                0, 0);
        }
        __builtin_amdgcn_s_wait_asynccnt(0);
    }
#else
    for (int idx = tid; idx < HMM_C * HMM_C; idx += 128) {
        const int i = idx >> 6, j = idx & 63;
        As[i * PSTR + j] = A16[idx];
    }
    {
        const uint4* gx = reinterpret_cast<const uint4*>(gxp);
        uint4* sx = reinterpret_cast<uint4*>(xs);
        for (int idx = tid; idx < LCH * (HMM_C / 4); idx += 128) sx[idx] = gx[idx];
    }
#endif
    __syncthreads();

    int exsum = 0;

    // ---- initialize P with the first factor of this chunk ----
    {
        float pmax = 0.f;
        for (int k = 0; k < 32; ++k) {
            const int idx = tid + 128 * k;
            const int i = idx >> 6, j = idx & 63;
            float v;
            if (c == 0) v = (i == j) ? pi[j] * __expf(xs[j]) : 0.f;
            else        v = (float)As[i * PSTR + j] * __expf(xs[j]);
            Ps[i * PSTR + j] = (_Float16)v;
            pmax = fmaxf(pmax, v);
        }
        #pragma unroll
        for (int off = 16; off > 0; off >>= 1)
            pmax = fmaxf(pmax, __shfl_xor(pmax, off, 32));
        if (lane == 0) wmax[w] = pmax;
        __syncthreads();
        const float am = fmaxf(fmaxf(wmax[0], wmax[1]), fmaxf(wmax[2], wmax[3]));
        int ex; (void)frexpf(am, &ex);
        exsum += ex;
        const float sc = ldexpf(1.0f, -ex);
        for (int k = 0; k < 32; ++k) {
            const int idx = tid + 128 * k;
            const int i = idx >> 6, j = idx & 63;
            Ps[i * PSTR + j] = (_Float16)((float)Ps[i * PSTR + j] * sc);
        }
        __syncthreads();
    }

    // ---- preload constant B-fragments of A (lanes = N, elements = K) ----
    Frag16 bfrag[2][4];
    #pragma unroll
    for (int kt = 0; kt < 2; ++kt) {
        #pragma unroll
        for (int nt = 0; nt < 4; ++nt) {
            #pragma unroll
            for (int e = 0; e < 16; ++e) {
                const int i = 32 * kt + 8 * hi + (e < 8 ? e : 8 + e);   // K row of A
                bfrag[kt][nt].h[e] = As[i * PSTR + 16 * nt + l15];
            }
        }
    }

    // ---- main sequential chain: 127 WMMA matmul steps ----
    const int m = 16 * w + l15;                  // this lane's M row (A-frag)
    const uint4* prow = reinterpret_cast<const uint4*>(&Ps[m * PSTR]);

    for (int s = 1; s < LCH; ++s) {
        // A-fragments of current P (lanes = M, packed K in dwords)
        Frag16 af[2];
        #pragma unroll
        for (int kt = 0; kt < 2; ++kt) {
            af[kt].q[0] = prow[4 * kt + hi];
            af[kt].q[1] = prow[4 * kt + hi + 2];
        }

        v8f   acc[4];
        float pmax = 0.f;
        #pragma unroll
        for (int nt = 0; nt < 4; ++nt) {
            v8f a = {0.f, 0.f, 0.f, 0.f, 0.f, 0.f, 0.f, 0.f};
            a = __builtin_amdgcn_wmma_f32_16x16x32_f16(
                    false, af[0].h, false, bfrag[0][nt].h, (short)0, a, false, false);
            a = __builtin_amdgcn_wmma_f32_16x16x32_f16(
                    false, af[1].h, false, bfrag[1][nt].h, (short)0, a, false, false);
            // column scale by exp(x_t[j]); D has N (=j) striped across lanes
            const float e = __expf(xs[(s << 6) + (nt << 4) + l15]);
            #pragma unroll
            for (int v = 0; v < 8; ++v) {
                a[v] *= e;
                pmax = fmaxf(pmax, a[v]);
            }
            acc[nt] = a;
        }

        #pragma unroll
        for (int off = 16; off > 0; off >>= 1)
            pmax = fmaxf(pmax, __shfl_xor(pmax, off, 32));
        if (lane == 0) wmax[w] = pmax;
        __syncthreads();   // all frag reads done + partial maxes visible
        const float am = fmaxf(fmaxf(wmax[0], wmax[1]), fmaxf(wmax[2], wmax[3]));
        int ex; (void)frexpf(am, &ex);
        exsum += ex;
        const float sc = ldexpf(1.0f, -ex);      // exact pow2 rescale

        // store normalized P back (D layout: lane -> N, vgpr -> M)
        #pragma unroll
        for (int nt = 0; nt < 4; ++nt) {
            #pragma unroll
            for (int v = 0; v < 8; ++v) {
                Ps[(16 * w + 8 * hi + v) * PSTR + (nt << 4) + l15] =
                    (_Float16)(acc[nt][v] * sc);
            }
        }
        __syncthreads();   // P fully written before next step's reads
    }

    // ---- emit chunk matrix (unpadded row-major f16) + log scale ----
    {
        _Float16* gPc = Pout + (size_t)(b * NCH + c) * (HMM_C * HMM_C);
#if HMM_ASYNC_ST
        // 8192 B = 512 x b128 from padded LDS rows to packed global rows.
        for (int idx = tid; idx < 512; idx += 128) {
            const int i = idx >> 3, q = idx & 7;
            __builtin_amdgcn_global_store_async_from_lds_b128(
                (AS_GLOBAL v4i*)((AS_GLOBAL char*)gPc + (size_t)idx * 16),
                (AS_LDS v4i*)((AS_LDS char*)Ps + (size_t)i * (PSTR * 2) + (size_t)q * 16),
                0, 0);
        }
        __builtin_amdgcn_s_wait_asynccnt(0);
#else
        unsigned int* gp = reinterpret_cast<unsigned int*>(gPc);
        const unsigned int* sp = reinterpret_cast<const unsigned int*>(Ps);
        for (int idx = tid; idx < HMM_C * (HMM_C / 2); idx += 128) {
            const int i = idx >> 5, jd = idx & 31;
            gp[idx] = sp[i * (PSTR / 2) + jd];
        }
#endif
        if (tid == 0) logSout[b * NCH + c] = (float)exsum * LN2F;
    }
}

// ---------------------------------------------------------------------------
// Kernel 2: per batch, u^T = 1^T * P_0 * ... * P_{NCH-1};  logp = log(sum u).
//   grid = B blocks of one wave (32 lanes); lane l owns columns l and l+32.
// ---------------------------------------------------------------------------
__global__ __launch_bounds__(32) void hmm_combine(const _Float16* __restrict__ Pall,
                                                  const float* __restrict__ logS,
                                                  float* __restrict__ out) {
    __shared__ __align__(16) float    u[HMM_C];
    __shared__ __align__(16) _Float16 Pc[HMM_C * HMM_C];

    const int b = blockIdx.x;
    const int l = threadIdx.x;

    u[l] = 1.f;
    u[l + 32] = 1.f;
    float logacc = 0.f;
    __syncthreads();

    for (int c = 0; c < NCH; ++c) {
        const uint4* gp = reinterpret_cast<const uint4*>(
            Pall + (size_t)(b * NCH + c) * (HMM_C * HMM_C));
        uint4* sp = reinterpret_cast<uint4*>(Pc);
        #pragma unroll
        for (int k = 0; k < 16; ++k) sp[l + 32 * k] = gp[l + 32 * k];
        __syncthreads();

        float u0 = 0.f, u1 = 0.f;
        for (int i = 0; i < HMM_C; ++i) {
            const float ui = u[i];
            u0 = fmaf(ui, (float)Pc[i * HMM_C + l],      u0);
            u1 = fmaf(ui, (float)Pc[i * HMM_C + l + 32], u1);
        }

        float pm = fmaxf(u0, u1);
        #pragma unroll
        for (int off = 16; off > 0; off >>= 1)
            pm = fmaxf(pm, __shfl_xor(pm, off, 32));
        int ex; (void)frexpf(pm, &ex);
        const float sc = ldexpf(1.0f, -ex);
        logacc += (float)ex * LN2F + logS[b * NCH + c];

        __syncthreads();          // all reads of u done before overwrite
        u[l]      = u0 * sc;
        u[l + 32] = u1 * sc;
        __syncthreads();
    }

    float s = u[l] + u[l + 32];
    #pragma unroll
    for (int off = 16; off > 0; off >>= 1) s += __shfl_xor(s, off, 32);
    if (l == 0) out[b] = logf(s) + logacc;
}

// ---------------------------------------------------------------------------
// Launch: prep -> chunk compose -> combine (stream-ordered).
// Scratch layout in d_ws (needs ~8.02 MB):
//   [0, 8192)                : A f16 (64x64)
//   [8192, 8448)             : pi f32 (64)
//   [8448, 8448 + 8 MiB)     : chunk matrices f16  [B][NCH][64][64]
//   [+8 MiB, +8 MiB + 4 KiB) : chunk log-scales f32 [B][NCH]
// ---------------------------------------------------------------------------
extern "C" void kernel_launch(void* const* d_in, const int* in_sizes, int n_in,
                              void* d_out, int out_size, void* d_ws, size_t ws_size,
                              hipStream_t stream) {
    const float* x     = (const float*)d_in[0];   // [B, T, C] f32
    const float* logA  = (const float*)d_in[1];   // [C, C]    f32
    const float* logpi = (const float*)d_in[2];   // [C]       f32
    float*       out   = (float*)d_out;           // [B]       f32

    char* ws = (char*)d_ws;
    _Float16* A16  = (_Float16*)(ws);
    float*    pi   = (float*)(ws + 8192);
    _Float16* Pall = (_Float16*)(ws + 8448);
    float*    logS = (float*)(ws + 8448 +
                              (size_t)HMM_B * NCH * HMM_C * HMM_C * sizeof(_Float16));

    hmm_prep<<<1, 64, 0, stream>>>(logA, logpi, A16, pi);
    hmm_chunks<<<dim3(NCH, HMM_B), 128, 0, stream>>>(x, A16, pi, Pall, logS);
    hmm_combine<<<HMM_B, 32, 0, stream>>>(Pall, logS, out);
}
